// MEMOIRWrapper_88029649699347
// MI455X (gfx1250) — compile-verified, matrix operand order checked by reading, not run on hardware
//
#include <hip/hip_runtime.h>
#include <hip/hip_bf16.h>
#include <stdint.h>

// ---------------- problem constants (from reference) ----------------
constexpr int BB   = 8;       // batch
constexpr int SS   = 4096;    // sequence
constexpr int DD   = 2048;    // feature dim (GEMM K)
constexpr int OO   = 2048;    // output dim (GEMM N)
constexpr int MM   = 128;     // saved masks
constexpr int KTOP = 256;     // top-k
constexpr float IRR = 0.3f;

typedef __attribute__((ext_vector_type(16))) __bf16 v16bf;
typedef __attribute__((ext_vector_type(8)))  float  v8f;

#ifndef __has_builtin
#define __has_builtin(x) 0
#endif

#if defined(__HIP_DEVICE_COMPILE__) && __has_builtin(__builtin_amdgcn_global_load_async_to_lds_b128)
#define USE_ASYNC_LDS 1
#else
#define USE_ASYNC_LDS 0
#endif

// ---------------- helpers ----------------
__device__ __forceinline__ unsigned int pk_bf16(float a, float b) {
  // round-to-nearest-even f32 -> bf16, packed pair (a lo 16, b hi 16)
  unsigned int ua = __float_as_uint(a);
  unsigned int ub = __float_as_uint(b);
  ua += 0x7FFFu + ((ua >> 16) & 1u);
  ub += 0x7FFFu + ((ub >> 16) & 1u);
  return (ua >> 16) | (ub & 0xFFFF0000u);
}

struct U8x32 { uint4 lo; uint4 hi; };

// bf16 fragment from LDS tile stored row-major, pitch 48 bf16 (96B, 16B-aligned).
// ISA 7.12.2 (16-bit): lanes 0-15 -> row = lane, K {0..7} U {16..23};
//                      lanes 16-31 -> row = lane-16, K {8..15} U {24..31}.
__device__ __forceinline__ v16bf load_frag(const unsigned short* base, int lane) {
  int r    = lane & 15;
  int kOff = (lane & 16) ? 8 : 0;
  const unsigned short* p = base + r * 48 + kOff;
  U8x32 t;
  t.lo = *(const uint4*)(p);        // K +0..+7
  t.hi = *(const uint4*)(p + 16);   // K +16..+23
  return __builtin_bit_cast(v16bf, t);
}

__device__ __forceinline__ void mma_step(const unsigned short* At,
                                         const unsigned short* Bt,
                                         int waveM, int waveN, int lane,
                                         v8f acc[4][2]) {
  const unsigned short* Asub = At + (waveM * 64) * 48;
  const unsigned short* Bsub = Bt + (waveN * 32) * 48;
  v16bf b0 = load_frag(Bsub, lane);
  v16bf b1 = load_frag(Bsub + 16 * 48, lane);
#pragma unroll
  for (int mt = 0; mt < 4; ++mt) {
    v16bf af = load_frag(Asub + mt * 16 * 48, lane);
    acc[mt][0] = __builtin_amdgcn_wmma_f32_16x16x32_bf16(
        false, af, false, b0, (short)0, acc[mt][0], false, false);
    acc[mt][1] = __builtin_amdgcn_wmma_f32_16x16x32_bf16(
        false, af, false, b1, (short)0, acc[mt][1], false, false);
  }
}

__device__ __forceinline__ void epilogue(float* __restrict__ out,
                                         const float* __restrict__ bias,
                                         int b, int bM0, int bN0,
                                         int waveM, int waveN, int lane,
                                         v8f acc[4][2]) {
  // C layout (ISA 7.12.2 32-bit C/D): lanes 0-15 -> N=lane, M=r;
  // lanes 16-31 -> N=lane-16, M=8+r.
  float* obase = out + (size_t)b * SS * OO;
  const int mAdd = (lane & 16) ? 8 : 0;
  const int nSub = lane & 15;
#pragma unroll
  for (int mt = 0; mt < 4; ++mt) {
#pragma unroll
    for (int nt = 0; nt < 2; ++nt) {
      int m0 = bM0 + waveM * 64 + mt * 16 + mAdd;
      int n0 = bN0 + waveN * 32 + nt * 16 + nSub;
      float bv = bias[n0];
      float* o = obase + (size_t)m0 * OO + n0;
#pragma unroll
      for (int r = 0; r < 8; ++r)
        o[(size_t)r * OO] = acc[mt][nt][r] + bv;
    }
  }
}

#if USE_ASYNC_LDS
// Builtin signature (from hipcc diagnostic): param0 is a NON-const
// 'int __attribute__((vector_size(16))) __device__ *'  (AS1 int4*).
typedef int v4i_vs __attribute__((vector_size(16)));
typedef __attribute__((address_space(1))) v4i_vs as1_v4i;
typedef __attribute__((address_space(3))) v4i_vs as3_v4i;

__device__ __forceinline__ void asy_cp(const unsigned short* g, unsigned short* l) {
  // global -> LDS, 16B per lane, tracked by ASYNCcnt (no VGPR round-trip)
  __builtin_amdgcn_global_load_async_to_lds_b128(
      (as1_v4i*)(unsigned long long)(const void*)g,
      (as3_v4i*)(unsigned int)(unsigned long long)(void*)l,
      0, 0);
}
#endif

__device__ __forceinline__ void wait_async_4() {
#if defined(__HIP_DEVICE_COMPILE__)
#if __has_builtin(__builtin_amdgcn_s_wait_asynccnt)
  __builtin_amdgcn_s_wait_asynccnt(4);
#else
  asm volatile("s_wait_asynccnt 0x4" ::: "memory");
#endif
#endif
}
__device__ __forceinline__ void wait_async_0() {
#if defined(__HIP_DEVICE_COMPILE__)
#if __has_builtin(__builtin_amdgcn_s_wait_asynccnt)
  __builtin_amdgcn_s_wait_asynccnt(0);
#else
  asm volatile("s_wait_asynccnt 0x0" ::: "memory");
#endif
#endif
}

// ---------------- kernel 1: partial masked sums over S chunks ----------------
__global__ void k_partial(const float* __restrict__ x,
                          const long long* __restrict__ boundaries,
                          float* __restrict__ partial) {
  const int b = blockIdx.y;
  const int chunk = blockIdx.z;                 // 8 chunks of 512
  const int d = blockIdx.x * 256 + threadIdx.x;
  long long bv = boundaries[b];
  int bc = (int)(bv < 0 ? 0 : (bv > (SS - 1) ? (SS - 1) : bv));
  const int CH = SS / 8;
  int sBeg = chunk * CH;
  int sEnd = sBeg + CH;
  int lim  = bc + 1;                             // positions s <= boundary
  if (sEnd > lim) sEnd = lim;
  float sum = 0.0f;
  const float* xp = x + ((size_t)b * SS + sBeg) * DD + d;
  for (int s = sBeg; s < sEnd; ++s, xp += DD) {
    __builtin_prefetch(xp + 16 * DD, 0, 1);     // global_prefetch_b8
    sum += *xp;
  }
  partial[((size_t)chunk * BB + b) * DD + d] = sum;
}

// ---------------- kernel 2: finalize aggregate (mean, decenter) ----------------
__global__ void k_agg(const float* __restrict__ partial,
                      const long long* __restrict__ boundaries,
                      const float* __restrict__ bg_mean,
                      float* __restrict__ agg) {
  const int b = blockIdx.y;
  const int d = blockIdx.x * 256 + threadIdx.x;
  long long bv = boundaries[b];
  int bc = (int)(bv < 0 ? 0 : (bv > (SS - 1) ? (SS - 1) : bv));
  float denom = (float)(bc + 1);
  if (denom < 1.0f) denom = 1.0f;
  float s = 0.0f;
  for (int c = 0; c < 8; ++c) s += partial[((size_t)c * BB + b) * DD + d];
  agg[(size_t)b * DD + d] = s / denom - bg_mean[d];
}

// ---------------- kernel 3: top-k by rank + permuted scatter ----------------
__global__ void k_topk(const float* __restrict__ agg,
                       const long long* __restrict__ perm,
                       unsigned char* __restrict__ active) {
  const int b = blockIdx.x;
  const int tid = threadIdx.x;                  // 256 threads
  __shared__ float sabs[DD];
  __shared__ unsigned char act[DD];
  for (int i = tid; i < DD; i += 256) {
    sabs[i] = fabsf(agg[(size_t)b * DD + i]);
    act[i] = 0;
  }
  __syncthreads();
  float myv[8];
  int cnt[8];
#pragma unroll
  for (int i = 0; i < 8; ++i) { myv[i] = sabs[tid + (i << 8)]; cnt[i] = 0; }
  for (int j = 0; j < DD; ++j) {
    float vj = sabs[j];
#pragma unroll
    for (int i = 0; i < 8; ++i) {
      int d = tid + (i << 8);
      // jax.lax.top_k: descending, ties broken by lower index
      cnt[i] += (vj > myv[i]) || ((vj == myv[i]) && (j < d));
    }
  }
#pragma unroll
  for (int i = 0; i < 8; ++i) {
    int d = tid + (i << 8);
    if (cnt[i] < KTOP) {
      int p = (int)perm[d];
      act[p] = 1;                                // LDS scatter (block-local)
    }
  }
  __syncthreads();
  for (int i = tid; i < DD; i += 256) active[(size_t)b * DD + i] = act[i];
}

// ---------------- kernel 4: overlap vs saved masks, argmax, final mask ----------------
__global__ void k_retrieve(const unsigned char* __restrict__ active,
                           const unsigned char* __restrict__ saved,
                           float* __restrict__ maskf) {
  const int b = blockIdx.x;
  const int tid = threadIdx.x;                  // 128 threads, one per saved mask
  __shared__ unsigned char act[DD];
  __shared__ int sov[MM];
  __shared__ int sidx[MM];
  __shared__ int sBest, sRel;
  for (int i = tid; i < DD; i += MM) act[i] = active[(size_t)b * DD + i];
  __syncthreads();
  int ov = 0;
  const unsigned char* sm = saved + (size_t)tid * DD;
  for (int d = 0; d < DD; ++d) ov += (int)(act[d] & sm[d]);
  sov[tid] = ov;
  sidx[tid] = tid;
  __syncthreads();
  for (int off = MM / 2; off > 0; off >>= 1) {
    if (tid < off) {
      int o2 = sov[tid + off], i2 = sidx[tid + off];
      if (o2 > sov[tid] || (o2 == sov[tid] && i2 < sidx[tid])) { // argmax -> first
        sov[tid] = o2; sidx[tid] = i2;
      }
    }
    __syncthreads();
  }
  if (tid == 0) {
    sBest = sidx[0];
    sRel = ((float)sov[0] * (1.0f / (float)KTOP) >= IRR) ? 1 : 0;
  }
  __syncthreads();
  const unsigned char* bm = saved + (size_t)sBest * DD;
  int rel = sRel;
  for (int d = tid; d < DD; d += MM)
    maskf[(size_t)b * DD + d] = (rel && bm[d]) ? 1.0f : 0.0f;
}

// ---------------- kernel 5a: x -> bf16 (one pass) ----------------
__global__ void k_cvt_x(const float* __restrict__ x, unsigned short* __restrict__ xh) {
  size_t i = ((size_t)blockIdx.x * 256 + threadIdx.x) * 8;
  float4 a = *(const float4*)(x + i);
  float4 c = *(const float4*)(x + i + 4);
  uint4 o;
  o.x = pk_bf16(a.x, a.y); o.y = pk_bf16(a.z, a.w);
  o.z = pk_bf16(c.x, c.y); o.w = pk_bf16(c.z, c.w);
  *(uint4*)(xh + i) = o;
}

// ---------------- kernel 5b: W_eff[b] = W + m_b * Wn -> bf16 ----------------
__global__ void k_cvt_w(const float* __restrict__ w, const float* __restrict__ nw,
                        const float* __restrict__ maskf, unsigned short* __restrict__ wc) {
  const int n = blockIdx.x;                     // output row
  const int b = blockIdx.y;                     // batch
  const int k = threadIdx.x * 8;                // 256 threads * 8 = 2048
  const float* wp = w  + (size_t)n * DD + k;
  const float* np = nw + (size_t)n * DD + k;
  const float* mp = maskf + (size_t)b * DD + k;
  uint4 o;
  float4 w0 = *(const float4*)(wp),     w1 = *(const float4*)(wp + 4);
  float4 n0 = *(const float4*)(np),     n1 = *(const float4*)(np + 4);
  float4 m0 = *(const float4*)(mp),     m1 = *(const float4*)(mp + 4);
  o.x = pk_bf16(w0.x + m0.x * n0.x, w0.y + m0.y * n0.y);
  o.y = pk_bf16(w0.z + m0.z * n0.z, w0.w + m0.w * n0.w);
  o.z = pk_bf16(w1.x + m1.x * n1.x, w1.y + m1.y * n1.y);
  o.w = pk_bf16(w1.z + m1.z * n1.z, w1.w + m1.w * n1.w);
  *(uint4*)(wc + ((size_t)b * OO + n) * DD + k) = o;
}

// ---------------- kernel 6: pre-converted GEMM (async LDS staging) ----------------
// out[b] = xh[b] @ wc[b]^T + bias.  128x128 tile, K-step 32, 8 wave32 (2x4),
// each wave 4x2 WMMA tiles.  Inner loop: zero conversion VALU.
#if USE_ASYNC_LDS
constexpr int NBUF = 2;
#else
constexpr int NBUF = 1;
#endif

__global__ __launch_bounds__(256, 1)
void k_gemm_pre(const unsigned short* __restrict__ xh,
                const unsigned short* __restrict__ wc,
                const float* __restrict__ bias,
                float* __restrict__ out) {
  const int b   = blockIdx.z;
  const int bN0 = blockIdx.x * 128;
  const int bM0 = blockIdx.y * 128;
  const int tid  = threadIdx.x;
  const int lane = tid & 31;
  const int wid  = tid >> 5;
  const int waveM = wid >> 2;                   // 0..1
  const int waveN = wid & 3;                    // 0..3

  __shared__ __align__(16) unsigned short At[NBUF * 128 * 48];
  __shared__ __align__(16) unsigned short Bt[NBUF * 128 * 48];

  // staging: each thread owns 32B (16 bf16) of one tile row
  const int row = tid >> 1;                     // 0..127
  const int qe  = (tid & 1) * 16;               // element offset in 32-elem row
  const unsigned short* ga = xh + ((size_t)b * SS + bM0 + row) * DD + qe;
  const unsigned short* gb = wc + ((size_t)b * OO + bN0 + row) * DD + qe;
  const int ldsOff = row * 48 + qe;

  v8f acc[4][2];
#pragma unroll
  for (int mt = 0; mt < 4; ++mt)
#pragma unroll
    for (int nt = 0; nt < 2; ++nt)
      acc[mt][nt] = (v8f){0.f, 0.f, 0.f, 0.f, 0.f, 0.f, 0.f, 0.f};

  const int KST = DD / 32;

#if USE_ASYNC_LDS
  // ---- double-buffered async global->LDS pipeline ----
  asy_cp(ga, &At[ldsOff]);           asy_cp(ga + 8, &At[ldsOff + 8]);
  asy_cp(gb, &Bt[ldsOff]);           asy_cp(gb + 8, &Bt[ldsOff + 8]);
  for (int ks = 0; ks < KST; ++ks) {
    const int cur = ks & 1;
    __syncthreads();                 // all waves done computing on buf[1-cur]
    if (ks + 1 < KST) {
      const int k1 = (ks + 1) * 32;
      const int no = (cur ^ 1) * 128 * 48 + ldsOff;
      asy_cp(ga + k1,     &At[no]);  asy_cp(ga + k1 + 8, &At[no + 8]);
      asy_cp(gb + k1,     &Bt[no]);  asy_cp(gb + k1 + 8, &Bt[no + 8]);
      wait_async_4();                // asyncs retire in order: buf[cur] complete
    } else {
      wait_async_0();
    }
    __syncthreads();                 // every wave's contribution to buf[cur] done
    mma_step(At + cur * 128 * 48, Bt + cur * 128 * 48, waveM, waveN, lane, acc);
  }
#else
  // ---- register-staged pipeline (no conversion, plain copies) ----
  uint4 ra0, ra1, rb0, rb1;
  ra0 = *(const uint4*)(ga);      ra1 = *(const uint4*)(ga + 8);
  rb0 = *(const uint4*)(gb);      rb1 = *(const uint4*)(gb + 8);
  for (int ks = 0; ks < KST; ++ks) {
    __syncthreads();
    *(uint4*)&At[ldsOff] = ra0;   *(uint4*)&At[ldsOff + 8] = ra1;
    *(uint4*)&Bt[ldsOff] = rb0;   *(uint4*)&Bt[ldsOff + 8] = rb1;
    __syncthreads();
    if (ks + 1 < KST) {
      const int k1 = (ks + 1) * 32;
      ra0 = *(const uint4*)(ga + k1);  ra1 = *(const uint4*)(ga + k1 + 8);
      rb0 = *(const uint4*)(gb + k1);  rb1 = *(const uint4*)(gb + k1 + 8);
    }
    mma_step(At, Bt, waveM, waveN, lane, acc);
  }
#endif

  epilogue(out, bias, b, bM0, bN0, waveM, waveN, lane, acc);
}

// ---------------- kernel 7: fully fused GEMM (fallback, small workspace) ----------------
__global__ __launch_bounds__(256, 1)
void k_gemm(const float* __restrict__ x,
            const float* __restrict__ weight,
            const float* __restrict__ bias,
            const float* __restrict__ new_weight,
            const float* __restrict__ maskf,
            float* __restrict__ out) {
  const int b   = blockIdx.z;
  const int bN0 = blockIdx.x * 128;
  const int bM0 = blockIdx.y * 128;
  const int tid  = threadIdx.x;
  const int lane = tid & 31;
  const int wid  = tid >> 5;
  const int waveM = wid >> 2;
  const int waveN = wid & 3;

  __shared__ __align__(16) unsigned short At[128 * 48];
  __shared__ __align__(16) unsigned short Bt[128 * 48];
  __shared__ float mLds[DD];

  for (int i = tid; i < DD; i += 256) mLds[i] = maskf[(size_t)b * DD + i];
  __syncthreads();

  const int r0 = tid >> 3;
  const int kq = (tid & 7) << 2;

  const float* xbase = x + ((size_t)b * SS + bM0 + r0) * DD + kq;
  const float* wbase = weight     + (size_t)(bN0 + r0) * DD + kq;
  const float* nbase = new_weight + (size_t)(bN0 + r0) * DD + kq;

  float4 ax[4], wx[4], nx[4];
#pragma unroll
  for (int i = 0; i < 4; ++i) {
    ax[i] = *(const float4*)(xbase + (size_t)(i * 32) * DD);
    wx[i] = *(const float4*)(wbase + (size_t)(i * 32) * DD);
    nx[i] = *(const float4*)(nbase + (size_t)(i * 32) * DD);
  }

  v8f acc[4][2];
#pragma unroll
  for (int mt = 0; mt < 4; ++mt)
#pragma unroll
    for (int nt = 0; nt < 2; ++nt)
      acc[mt][nt] = (v8f){0.f, 0.f, 0.f, 0.f, 0.f, 0.f, 0.f, 0.f};

  const int KSTEPS = DD / 32;
  for (int ks = 0; ks < KSTEPS; ++ks) {
    const int k0 = ks * 32;
    __syncthreads();
#pragma unroll
    for (int i = 0; i < 4; ++i) {
      int r = r0 + i * 32;
      *(uint2*)&At[r * 48 + kq] =
          make_uint2(pk_bf16(ax[i].x, ax[i].y), pk_bf16(ax[i].z, ax[i].w));
      float m0 = mLds[k0 + kq + 0], m1 = mLds[k0 + kq + 1];
      float m2 = mLds[k0 + kq + 2], m3 = mLds[k0 + kq + 3];
      float c0 = wx[i].x + m0 * nx[i].x;
      float c1 = wx[i].y + m1 * nx[i].y;
      float c2 = wx[i].z + m2 * nx[i].z;
      float c3 = wx[i].w + m3 * nx[i].w;
      *(uint2*)&Bt[r * 48 + kq] = make_uint2(pk_bf16(c0, c1), pk_bf16(c2, c3));
    }
    __syncthreads();
    if (ks + 1 < KSTEPS) {
      const int k1 = (ks + 1) * 32;
#pragma unroll
      for (int i = 0; i < 4; ++i) {
        ax[i] = *(const float4*)(xbase + (size_t)(i * 32) * DD + k1);
        wx[i] = *(const float4*)(wbase + (size_t)(i * 32) * DD + k1);
        nx[i] = *(const float4*)(nbase + (size_t)(i * 32) * DD + k1);
      }
    }
    mma_step(At, Bt, waveM, waveN, lane, acc);
  }

  epilogue(out, bias, b, bM0, bN0, waveM, waveN, lane, acc);
}

// ---------------- launch ----------------
extern "C" void kernel_launch(void* const* d_in, const int* in_sizes, int n_in,
                              void* d_out, int out_size, void* d_ws, size_t ws_size,
                              hipStream_t stream) {
  const float*      x          = (const float*)d_in[0];
  const long long*  boundaries = (const long long*)d_in[1];
  const float*      weight     = (const float*)d_in[2];
  const float*      bias       = (const float*)d_in[3];
  const float*      new_weight = (const float*)d_in[4];
  const long long*  permut     = (const long long*)d_in[5];
  const unsigned char* saved   = (const unsigned char*)d_in[6];
  const float*      bg_mean    = (const float*)d_in[7];
  // d_in[8] = top_k scalar (256, hardcoded)
  (void)in_sizes; (void)n_in; (void)out_size;

  // small workspace region (< 1 MB)
  char* ws = (char*)d_ws;
  float* partial = (float*)ws;                               // 512 KB
  float* agg     = partial + (size_t)8 * BB * DD;            // 64 KB
  float* maskf   = agg + (size_t)BB * DD;                    // 64 KB
  unsigned char* active = (unsigned char*)(maskf + (size_t)BB * DD); // 16 KB

  // large (optional) region: bf16 x and per-batch combined bf16 weights
  const size_t SMALL = (size_t)1 << 20;
  const size_t XH_BYTES = (size_t)BB * SS * DD * 2;          // 128 MB
  const size_t WC_BYTES = (size_t)BB * OO * DD * 2;          // 64 MB
  const size_t NEED = SMALL + XH_BYTES + WC_BYTES;
  unsigned short* xh = (unsigned short*)(ws + SMALL);
  unsigned short* wc = (unsigned short*)(ws + SMALL + XH_BYTES);

  float* out = (float*)d_out;

  k_partial <<<dim3(DD / 256, BB, 8), 256, 0, stream>>>(x, boundaries, partial);
  k_agg     <<<dim3(DD / 256, BB),    256, 0, stream>>>(partial, boundaries, bg_mean, agg);
  k_topk    <<<BB, 256, 0, stream>>>(agg, permut, active);
  k_retrieve<<<BB, MM,  0, stream>>>(active, saved, maskf);

  if (ws_size >= NEED) {
    // fast path: hoist all f32->bf16 conversion + weight combine out of GEMM
    const int nCvtX = (int)(((size_t)BB * SS * DD) / (256 * 8));
    k_cvt_x <<<nCvtX, 256, 0, stream>>>(x, xh);
    k_cvt_w <<<dim3(OO, BB), 256, 0, stream>>>(weight, new_weight, maskf, wc);
    k_gemm_pre<<<dim3(OO / 128, SS / 128, BB), 256, 0, stream>>>(xh, wc, bias, out);
  } else {
    // fallback: fully fused (conversion in the GEMM loop)
    k_gemm<<<dim3(OO / 128, SS / 128, BB), 256, 0, stream>>>(
        x, weight, bias, new_weight, maskf, out);
  }
}
